// wAttention_81827716923995
// MI455X (gfx1250) — matrix operand order repaired and checked by reading, not compiled, not run on hardware
//
#include <hip/hip_runtime.h>

typedef __bf16 bf16;
typedef __attribute__((ext_vector_type(16))) __bf16 v16bf;
typedef __attribute__((ext_vector_type(4)))  __bf16 v4bf;
typedef __attribute__((ext_vector_type(8)))  float  v8f;

#define N_ROWS 8192
#define M_KEYS 8192
#define D_DIM  128
#define BK     32
#define WAVES  4
#define ROWS_PER_BLOCK (WAVES * 16)   // 64
#define LDSK_STRIDE  136              // 128 + 8 pad (bf16): 272B rows -> conflict-free b128 reads
#define LDSKT_STRIDE 40               // 32 + 8 pad  (bf16): 80B rows
#define LDSP_STRIDE  40
#define LDSK_BUF  (BK * LDSK_STRIDE)
#define LDSKT_BUF (D_DIM * LDSKT_STRIDE)

#define NEG_BIG (-9.0e15f)
#define SLOPE   0.2f

union FragA {
    v16bf v;
    bf16  e[16];
    uint4 q[2];
};

__device__ __forceinline__ float hmax16(float v) {
    #pragma unroll
    for (int m = 8; m >= 1; m >>= 1) v = fmaxf(v, __shfl_xor(v, m, 32));
    return v;
}
__device__ __forceinline__ float hsum16(float v) {
    #pragma unroll
    for (int m = 8; m >= 1; m >>= 1) v += __shfl_xor(v, m, 32);
    return v;
}

__global__ __launch_bounds__(WAVES * 32)
void wattn_flash_kernel(const float* __restrict__ qs,
                        const float* __restrict__ ks,
                        const int*   __restrict__ mask,
                        float*       __restrict__ out)
{
    // double-buffered K tile (two layouts) + per-wave P restage scratch
    __shared__ __align__(16) bf16 ldsK [2 * LDSK_BUF];
    __shared__ __align__(16) bf16 ldsKT[2 * LDSKT_BUF];
    __shared__ __align__(16) bf16 ldsP [WAVES * 16 * LDSP_STRIDE];

    const int tid   = threadIdx.x;
    const int wave  = tid >> 5;
    const int lane  = tid & 31;
    const int lhalf = lane >> 4;
    const int lmod  = lane & 15;
    const int qRow0 = blockIdx.x * ROWS_PER_BLOCK + wave * 16;

    // ---- Q A-fragments (bf16), 4 chunks of K=32 over D=128 ----
    // A operand (16x32), contiguous k order used here (B matches): lanes 0-15 row=lmod hold K 0..15; lanes 16-31 hold K 16..31
    FragA qf[4];
    {
        const float* qrow = qs + (size_t)(qRow0 + lmod) * D_DIM;
        #pragma unroll
        for (int c = 0; c < 4; ++c) {
            const int f0 = c * 32 + lhalf * 16;
            const int f1 = c * 32 + lhalf * 16 + 8;
            #pragma unroll
            for (int i = 0; i < 8; ++i) qf[c].e[i]     = (bf16)qrow[f0 + i];
            #pragma unroll
            for (int i = 0; i < 8; ++i) qf[c].e[8 + i] = (bf16)qrow[f1 + i];
        }
    }

    // ---- accumulators / softmax state ----
    v8f   oacc[8];
    float mrun[8], lrun[8];
    #pragma unroll
    for (int f = 0; f < 8; ++f) oacc[f] = (v8f){0,0,0,0,0,0,0,0};
    #pragma unroll
    for (int r = 0; r < 8; ++r) { mrun[r] = -__builtin_huge_valf(); lrun[r] = 0.0f; }

    // ---- K-tile register prefetch: each thread owns 128B contiguous of one key row ----
    const int skey = tid >> 2;            // 0..31
    const int sfb  = (tid & 3) * 32;      // 0,32,64,96
    float4 kreg[8];
    {
        const float4* krow = (const float4*)(ks + (size_t)skey * D_DIM + sfb);
        #pragma unroll
        for (int j = 0; j < 8; ++j) kreg[j] = krow[j];
    }

    const int numTiles = M_KEYS / BK;
    for (int kt = 0; kt < numTiles; ++kt) {
        const int buf = kt & 1;
        bf16* dK  = ldsK  + buf * LDSK_BUF;
        bf16* dKT = ldsKT + buf * LDSKT_BUF;

        // ---- stage prefetched tile: fp32 -> bf16, key-major + feature-major ----
        #pragma unroll
        for (int j = 0; j < 8; ++j) {
            float4 x = kreg[j];
            const int f = sfb + j * 4;
            bf16 b0 = (bf16)x.x, b1 = (bf16)x.y, b2 = (bf16)x.z, b3 = (bf16)x.w;
            *(v4bf*)&dK[skey * LDSK_STRIDE + f] = (v4bf){b0, b1, b2, b3};
            dKT[(f + 0) * LDSKT_STRIDE + skey] = b0;
            dKT[(f + 1) * LDSKT_STRIDE + skey] = b1;
            dKT[(f + 2) * LDSKT_STRIDE + skey] = b2;
            dKT[(f + 3) * LDSKT_STRIDE + skey] = b3;
        }
        __syncthreads();   // single barrier per tile: stores visible; prior-buffer reads proven done

        // ---- issue next tile's global loads; latency hidden by compute below ----
        if (kt + 1 < numTiles) {
            const float4* krow =
                (const float4*)(ks + (size_t)((kt + 1) * BK + skey) * D_DIM + sfb);
            #pragma unroll
            for (int j = 0; j < 8; ++j) kreg[j] = krow[j];
        }

        // ---- preload all 8 QK^T B-fragments, then run the WMMA chain ----
        // B operand (32x16), same contiguous k order as Q: lane col n=lmod; lanes 0-15 K=0..15, lanes 16-31 K=16..31
        FragA bq[8];
        #pragma unroll
        for (int c = 0; c < 4; ++c) {
            const int fbase = c * 32 + lhalf * 16;
            bq[2*c+0].q[0] = *(const uint4*)&dK[lmod        * LDSK_STRIDE + fbase];
            bq[2*c+0].q[1] = *(const uint4*)&dK[lmod        * LDSK_STRIDE + fbase + 8];
            bq[2*c+1].q[0] = *(const uint4*)&dK[(16 + lmod) * LDSK_STRIDE + fbase];
            bq[2*c+1].q[1] = *(const uint4*)&dK[(16 + lmod) * LDSK_STRIDE + fbase + 8];
        }
        v8f s0 = (v8f){0,0,0,0,0,0,0,0};
        v8f s1 = (v8f){0,0,0,0,0,0,0,0};
        #pragma unroll
        for (int c = 0; c < 4; ++c) {
            s0 = __builtin_amdgcn_wmma_f32_16x16x32_bf16(false, qf[c].v, false, bq[2*c+0].v, (short)0, s0, false, false);
            s1 = __builtin_amdgcn_wmma_f32_16x16x32_bf16(false, qf[c].v, false, bq[2*c+1].v, (short)0, s1, false, false);
        }

        // ---- leaky-relu + mask (stream-once int32: non-temporal) ----
        float p0[8], p1[8];
        const int colBase = kt * BK + lmod;
        #pragma unroll
        for (int r = 0; r < 8; ++r) {
            const int row = qRow0 + r + lhalf * 8;   // C layout: vgpr r holds row r + 8*half
            const size_t mb = (size_t)row * M_KEYS + colBase;
            const int mk0 = __builtin_nontemporal_load(&mask[mb]);
            const int mk1 = __builtin_nontemporal_load(&mask[mb + 16]);
            float x0 = s0[r]; x0 = x0 > 0.0f ? x0 : SLOPE * x0;
            float x1 = s1[r]; x1 = x1 > 0.0f ? x1 : SLOPE * x1;
            p0[r] = mk0 > 0 ? x0 : NEG_BIG;
            p1[r] = mk1 > 0 ? x1 : NEG_BIG;
        }

        // ---- preload V B-fragments early: LDS latency hides under softmax VALU ----
        FragA vfr[8];
        #pragma unroll
        for (int f = 0; f < 8; ++f) {
            const bf16* vrow = dKT + (f * 16 + lmod) * LDSKT_STRIDE + lhalf * 16;
            vfr[f].q[0] = ((const uint4*)vrow)[0];
            vfr[f].q[1] = ((const uint4*)vrow)[1];
        }

        // ---- online softmax update ----
        #pragma unroll
        for (int r = 0; r < 8; ++r) {
            const float tmax  = hmax16(fmaxf(p0[r], p1[r]));
            const float mn    = fmaxf(mrun[r], tmax);
            const float scale = __expf(mrun[r] - mn);
            const float e0 = __expf(p0[r] - mn);
            const float e1 = __expf(p1[r] - mn);
            lrun[r] = lrun[r] * scale + hsum16(e0 + e1);
            mrun[r] = mn;
            p0[r] = e0; p1[r] = e1;
            #pragma unroll
            for (int f = 0; f < 8; ++f) oacc[f][r] *= scale;
        }

        // ---- restage P: C layout -> A layout via per-wave LDS scratch ----
        bf16* pw = ldsP + wave * 16 * LDSP_STRIDE;
        #pragma unroll
        for (int r = 0; r < 8; ++r) {
            const int row = r + lhalf * 8;
            pw[row * LDSP_STRIDE + lmod]      = (bf16)p0[r];
            pw[row * LDSP_STRIDE + 16 + lmod] = (bf16)p1[r];
        }
        asm volatile("s_wait_dscnt 0" ::: "memory");   // wave-local producer->consumer
        FragA pf;
        {
            const bf16* prow = pw + lmod * LDSP_STRIDE;
            pf.q[0] = *(const uint4*)&prow[lhalf * 16];
            pf.q[1] = *(const uint4*)&prow[lhalf * 16 + 8];
        }

        // ---- O += P * V  (V == K, feature-major tile) ----
        #pragma unroll
        for (int f = 0; f < 8; ++f) {
            oacc[f] = __builtin_amdgcn_wmma_f32_16x16x32_bf16(false, pf.v, false, vfr[f].v, (short)0, oacc[f], false, false);
        }
    }

    // ---- epilogue: normalize and store ----
    #pragma unroll
    for (int r = 0; r < 8; ++r) {
        const int row = qRow0 + r + lhalf * 8;
        const float inv = 1.0f / lrun[r];
        #pragma unroll
        for (int f = 0; f < 8; ++f) {
            out[(size_t)row * D_DIM + f * 16 + lmod] = oacc[f][r] * inv;
        }
    }
}

extern "C" void kernel_launch(void* const* d_in, const int* in_sizes, int n_in,
                              void* d_out, int out_size, void* d_ws, size_t ws_size,
                              hipStream_t stream) {
    const float* qs   = (const float*)d_in[0];
    const float* ks   = (const float*)d_in[1];
    const int*   mask = (const int*)d_in[2];
    float*       out  = (float*)d_out;

    dim3 grid(N_ROWS / ROWS_PER_BLOCK);   // 128 blocks
    dim3 block(WAVES * 32);               // 128 threads = 4 waves
    wattn_flash_kernel<<<grid, block, 0, stream>>>(qs, ks, mask, out);
}
